// LSTM_3659312136319
// MI455X (gfx1250) — compile-verified, matrix operand order checked by reading, not run on hardware
//
#include <hip/hip_runtime.h>
#include <hip/hip_bf16.h>

typedef __attribute__((ext_vector_type(16))) _Float16 v16h;
typedef __attribute__((ext_vector_type(8)))  _Float16 v8h;
typedef __attribute__((ext_vector_type(8)))  float    v8f;

static constexpr int Bsz = 4096;
static constexpr int Tsz = 512;
static constexpr int Hsz = 64;       // hidden
static constexpr int ROWS = 16;      // batch rows per workgroup (one WMMA M tile)
static constexpr int HS_STRIDE = 72; // padded h row stride (halves): 144B rows

__device__ __forceinline__ float fast_sigmoid(float x) {
  // 1 / (1 + 2^(-x*log2e))
  return __builtin_amdgcn_rcpf(1.0f + __builtin_amdgcn_exp2f(-1.4426950408889634f * x));
}
__device__ __forceinline__ float fast_tanh(float x) {
  // 2*sigmoid(2x) - 1
  return 2.0f * __builtin_amdgcn_rcpf(1.0f + __builtin_amdgcn_exp2f(-2.885390081777927f * x)) - 1.0f;
}

__global__ __launch_bounds__(128) void lstm_wmma_kernel(
    const float* __restrict__ x,     // [B, T]  (I==1)
    const float* __restrict__ w_ih,  // [256]   (I==1)
    const float* __restrict__ w_hh,  // [256, 64]
    const float* __restrict__ b_ih,  // [256]
    const float* __restrict__ b_hh,  // [256]
    const float* __restrict__ w_out, // [64]    (O==1)
    const float* __restrict__ b_out, // [1]
    float* __restrict__ out)         // [B]
{
  __shared__ __align__(16) float    xs[Tsz * ROWS];            // 32 KB: xs[t*16+m]
  __shared__ __align__(16) _Float16 hs[2][ROWS * HS_STRIDE];   // 4.5 KB: double-buffered h (f16)

  const int tid  = threadIdx.x;
  const int wv   = tid >> 5;    // wave 0..3 : hidden slice [16*wv, 16*wv+16)
  const int lane = tid & 31;
  const int r0   = blockIdx.x * ROWS;

  // ---- one-time: stage x tile (coalesced HBM reads), zero both h buffers ----
  for (int i = tid; i < ROWS * Tsz; i += 128) {
    int m = i >> 9;            // row (Tsz==512)
    int t = i & (Tsz - 1);
    xs[t * ROWS + m] = x[(size_t)(r0 + m) * Tsz + t];
  }
  {
    _Float16* hflat = &hs[0][0];
    for (int i = tid; i < 2 * ROWS * HS_STRIDE; i += 128) hflat[i] = (_Float16)0.0f;
  }

  // ---- one-time: preload B = w_hh^T tiles into registers (f32 -> f16) ----
  // Wave wv owns, for all four gates g, the 16 gate-columns [g*64 + 16*wv, +16).
  // 16-bit B 32x16 layout: lane n(<16) holds col N=n with K{0..7,16..23};
  // lane n+16 holds the same column with K{8..15,24..31}.
  const int n     = lane & 15;
  const int kb    = (lane >> 4) * 8;  // K sub-base within a 32-chunk
  const int hbase = wv * 16;          // hidden-slice base column
  v16h Breg[4][2];
  float wih[4], bias[4];
  #pragma unroll
  for (int g = 0; g < 4; ++g) {
    const int col = g * 64 + hbase + n;     // gate column 0..255 (i,f,g,o blocks)
    const float* wr = w_hh + col * Hsz;     // row of w_hh = column of B
    #pragma unroll
    for (int q = 0; q < 2; ++q) {
      #pragma unroll
      for (int i = 0; i < 8; ++i) {
        Breg[g][q][i]     = (_Float16)wr[q * 32 + kb + i];
        Breg[g][q][i + 8] = (_Float16)wr[q * 32 + kb + 16 + i];
      }
    }
    wih[g]  = w_ih[col];
    bias[g] = b_ih[col] + b_hh[col];
  }

  // A layout: lane<16 -> M=lane, K base 0; lane>=16 -> M=lane-16, K base 8
  const int arow  = lane & 15;
  const int mbase = (lane >> 4) * 8;  // C/D layout: VGPR i holds row i+mbase, col n

  v8f c;                              // cell state for (rows mbase..mbase+7, col hbase+n)
  #pragma unroll
  for (int i = 0; i < 8; ++i) c[i] = 0.0f;

  __syncthreads();

  #pragma unroll 1
  for (int t = 0; t < Tsz; ++t) {
    const int rb = t & 1;             // read buffer (h_{t})
    const int wb = rb ^ 1;            // write buffer (h_{t+1})

    // ---- load h_t in WMMA A layout (two K-chunks), 16B LDS loads ----
    const v8h l0 = *(const v8h*)&hs[rb][arow * HS_STRIDE + kb];
    const v8h l1 = *(const v8h*)&hs[rb][arow * HS_STRIDE + kb + 16];
    const v8h l2 = *(const v8h*)&hs[rb][arow * HS_STRIDE + kb + 32];
    const v8h l3 = *(const v8h*)&hs[rb][arow * HS_STRIDE + kb + 48];
    v16h a_lo, a_hi;
    #pragma unroll
    for (int i = 0; i < 8; ++i) {
      a_lo[i] = l0[i]; a_lo[i + 8] = l1[i];
      a_hi[i] = l2[i]; a_hi[i + 8] = l3[i];
    }

    // ---- all four gate tiles for this wave's hidden slice ----
    // gates = (x*w_ih + bias) + h @ w_hh^T ; four independent 2-deep WMMA chains
    v8f acc[4];
    #pragma unroll
    for (int g = 0; g < 4; ++g) {
      v8f cinit;
      #pragma unroll
      for (int i = 0; i < 8; ++i)
        cinit[i] = xs[t * ROWS + mbase + i] * wih[g] + bias[g];  // vectorized LDS read
      v8f d = __builtin_amdgcn_wmma_f32_16x16x32_f16(
                  false, a_lo, false, Breg[g][0], (short)0, cinit, false, false);
      d     = __builtin_amdgcn_wmma_f32_16x16x32_f16(
                  false, a_hi, false, Breg[g][1], (short)0, d, false, false);
      acc[g] = d;
    }

    // ---- cell update entirely in registers (D layout matches across gates) ----
    #pragma unroll
    for (int i = 0; i < 8; ++i) {
      const float iv = fast_sigmoid(acc[0][i]);
      const float fv = fast_sigmoid(acc[1][i]);
      const float gv = fast_tanh   (acc[2][i]);
      const float ov = fast_sigmoid(acc[3][i]);
      c[i] = fv * c[i] + iv * gv;
      const float h = ov * fast_tanh(c[i]);
      hs[wb][(mbase + i) * HS_STRIDE + hbase + n] = (_Float16)h;
    }

    __syncthreads();   // single barrier per step (hs is double-buffered)
  }

  // ---- final projection on h_T: out[b] = h . w_out + b_out ----
  if (tid < ROWS) {
    const _Float16* hT = &hs[Tsz & 1][tid * HS_STRIDE];
    float s = b_out[0];
    #pragma unroll
    for (int j = 0; j < Hsz; ++j)
      s += (float)hT[j] * w_out[j];
    out[r0 + tid] = s;
  }
}

extern "C" void kernel_launch(void* const* d_in, const int* in_sizes, int n_in,
                              void* d_out, int out_size, void* d_ws, size_t ws_size,
                              hipStream_t stream) {
  const float* x     = (const float*)d_in[0];
  const float* w_ih  = (const float*)d_in[1];
  const float* w_hh  = (const float*)d_in[2];
  const float* b_ih  = (const float*)d_in[3];
  const float* b_hh  = (const float*)d_in[4];
  const float* w_out = (const float*)d_in[5];
  const float* b_out = (const float*)d_in[6];
  float* out = (float*)d_out;

  dim3 grid(Bsz / ROWS);   // 256 workgroups, each owns 16 batch rows for all 512 steps
  dim3 block(128);         // 4 wave32s: one hidden slice (all 4 gates) per wave
  hipLaunchKernelGGL(lstm_wmma_kernel, grid, block, 0, stream,
                     x, w_ih, w_hh, b_ih, b_hh, w_out, b_out, out);
}
